// GATGraphClassifier4HiddenLayers_14551349198963
// MI455X (gfx1250) — matrix operand order repaired and checked by reading, not compiled, not run on hardware
//
#include <hip/hip_runtime.h>

#define G      32
#define NN     1200
#define EE     19200
#define IN_DIM 128
#define HIDW   256
#define SUMH   1024
#define OUT_H  30
#define C2D    3
#define KK     13
#define POOL   40

typedef _Float16 v16h __attribute__((ext_vector_type(16)));
typedef _Float16 v8h  __attribute__((ext_vector_type(8)));
typedef float    v8f  __attribute__((ext_vector_type(8)));

// ---------------------------------------------------------------- utilities
__global__ void k_zero(float* __restrict__ p, int n) {
    int i = blockIdx.x * 256 + threadIdx.x;
    if (i < n) p[i] = 0.0f;
}

// out_deg[g,src[e]] += 1 ; in_deg[g,dst[e]] += 1
__global__ void k_degree(const int* __restrict__ src, const int* __restrict__ dst,
                         float* __restrict__ degOut, float* __restrict__ degIn) {
    int i = blockIdx.x * 256 + threadIdx.x;
    if (i >= G * EE) return;
    int g = i / EE;
    atomicAdd(&degOut[g * NN + src[i]], 1.0f);
    atomicAdd(&degIn [g * NN + dst[i]], 1.0f);
}

// deg -> rsqrt(clip(deg,1,inf)) in place, both arrays
__global__ void k_rinv(float* __restrict__ degOut, float* __restrict__ degIn) {
    int i = blockIdx.x * 256 + threadIdx.x;
    if (i >= G * NN) return;
    float a = degOut[i]; degOut[i] = rsqrtf(a < 1.0f ? 1.0f : a);
    float b = degIn[i];  degIn[i]  = rsqrtf(b < 1.0f ? 1.0f : b);
}

// W (K x 256, f32 row-major) -> Wt (256 x K, f16)  : makes WMMA B loads contiguous
__global__ void k_prep_w(const float* __restrict__ W, _Float16* __restrict__ Wt, int Kd) {
    int i = blockIdx.x * 256 + threadIdx.x;
    if (i >= Kd * HIDW) return;
    int k = i / HIDW, n = i - k * HIDW;          // coalesced read of W
    Wt[(size_t)n * Kd + k] = (_Float16)W[i];
}

// ---------------------------------------------------------------- WMMA GEMM
// X[g] = (in[g] * rinvOut[g][:,None]) @ W   ->  xtmp (N x 256 f32 per graph)
// block: 256 thr = 8 waves; block covers one 16-row M tile x all 256 output cols.
__global__ __launch_bounds__(256) void k_gemm(
        const float* __restrict__ in, int inStride, int colOff,
        const float* __restrict__ rinvOut, const _Float16* __restrict__ Wt,
        int Kd, float* __restrict__ xtmp) {
    __shared__ _Float16 As[16 * 256];            // 8 KB max
    const int g = blockIdx.y, mt = blockIdx.x;
    const int tid = threadIdx.x;

    // stage scaled A strip (16 rows x Kd) as f16, coalesced along k
    for (int idx = tid; idx < 16 * Kd; idx += 256) {
        int r = idx / Kd, k = idx - r * Kd;
        int row = mt * 16 + r;
        float v = in[(size_t)(g * NN + row) * inStride + colOff + k] * rinvOut[g * NN + row];
        As[r * Kd + k] = (_Float16)v;
    }
    __syncthreads();

    const int lane = tid & 31, wave = tid >> 5;
    const int half = lane >> 4, lm = lane & 15;  // ISA 16-bit A/B lane layout
    v8f c0 = {}, c1 = {};
    const int n0 = wave * 32, n1 = n0 + 16;      // 8 waves x 2 tiles = 256 cols
    const _Float16* ap  = As + lm * Kd + half * 16;
    const _Float16* bp0 = Wt + (size_t)(n0 + lm) * Kd + half * 16;
    const _Float16* bp1 = Wt + (size_t)(n1 + lm) * Kd + half * 16;

    for (int k0 = 0; k0 < Kd; k0 += 32) {
        v16h a = *(const v16h*)(ap + k0);        // K = k0 + half*16 + [0..15]
        v16h b0 = *(const v16h*)(bp0 + k0);      // K = k0 + half*16 + [0..15]
        v16h b1 = *(const v16h*)(bp1 + k0);
        c0 = __builtin_amdgcn_wmma_f32_16x16x32_f16(false, a, false, b0, (short)0, c0, false, false);
        c1 = __builtin_amdgcn_wmma_f32_16x16x32_f16(false, a, false, b1, (short)0, c1, false, false);
    }

    // D layout: lane holds col n+lm, rows (half*8 + r)
    float* ob = xtmp + (size_t)(g * NN + mt * 16 + half * 8) * HIDW;
    #pragma unroll
    for (int r = 0; r < 8; ++r) {
        ob[(size_t)r * HIDW + n0 + lm] = c0[r];
        ob[(size_t)r * HIDW + n1 + lm] = c1[r];
    }
}

// ---------------------------------------------------------------- scatter-add
__global__ void k_scatter(const float4* __restrict__ xtmp, const int* __restrict__ src,
                          const int* __restrict__ dst, float* __restrict__ agg) {
    int i = blockIdx.x * 256 + threadIdx.x;
    if (i >= G * EE * 64) return;
    int c4 = i & 63;
    int t  = i >> 6;
    int e  = t % EE, g = t / EE;
    int s = src[g * EE + e], d = dst[g * EE + e];
    float4 v = xtmp[(size_t)(g * NN + s) * 64 + c4];
    float* ap = agg + (size_t)(g * NN + d) * 256 + c4 * 4;
    atomicAdd(ap + 0, v.x); atomicAdd(ap + 1, v.y);
    atomicAdd(ap + 2, v.z); atomicAdd(ap + 3, v.w);
}

// agg * rinvIn + b, relu -> h_cat slab at column colOff
__global__ void k_finalize(const float4* __restrict__ agg, const float* __restrict__ rinvIn,
                           const float* __restrict__ bias, float* __restrict__ hcat, int colOff) {
    int i = blockIdx.x * 256 + threadIdx.x;
    if (i >= G * NN * 64) return;
    int c4 = i & 63;
    int t  = i >> 6;                 // = g*NN + node
    float4 v = agg[(size_t)i];
    float r  = rinvIn[t];
    float4 b = ((const float4*)bias)[c4];
    float4 o;
    o.x = fmaxf(v.x * r + b.x, 0.0f);
    o.y = fmaxf(v.y * r + b.y, 0.0f);
    o.z = fmaxf(v.z * r + b.z, 0.0f);
    o.w = fmaxf(v.w * r + b.w, 0.0f);
    *(float4*)(hcat + (size_t)t * SUMH + colOff + c4 * 4) = o;
}

// ---------------------------------------------------------------- conv13x13 + maxpool(40 rows)
// block: (g, pooled row p, 128-col tile). LDS tile 52x140, fused max over 40 conv rows.
__global__ __launch_bounds__(256) void k_convpool(
        const float* __restrict__ hcat, const float* __restrict__ cw,
        const float* __restrict__ cb, float* __restrict__ out) {
    __shared__ float tile[52 * 140];
    __shared__ float wl[C2D * KK * KK];
    __shared__ float red[2][C2D][128];
    const int g = blockIdx.z, p = blockIdx.y, xt = blockIdx.x;
    const int tid = threadIdx.x;

    for (int i = tid; i < C2D * KK * KK; i += 256) wl[i] = cw[i];
    const int row0 = p * POOL - (KK / 2), col0 = xt * 128 - (KK / 2);
    for (int i = tid; i < 52 * 140; i += 256) {
        int r = i / 140, c = i - r * 140;
        int gr = row0 + r, gc = col0 + c;
        float v = 0.0f;
        if (gr >= 0 && gr < NN && gc >= 0 && gc < SUMH)
            v = hcat[(size_t)(g * NN + gr) * SUMH + gc];
        tile[i] = v;
    }
    __syncthreads();

    const int col = tid & 127, half = tid >> 7;  // 2 halves x 20 conv rows each
    float m0 = -3.4e38f, m1 = -3.4e38f, m2 = -3.4e38f;
    for (int yg = 0; yg < 5; ++yg) {
        float acc[4][3];
        #pragma unroll
        for (int r = 0; r < 4; ++r) { acc[r][0] = 0.f; acc[r][1] = 0.f; acc[r][2] = 0.f; }
        const int ybase = half * 20 + yg * 4;
        for (int dy = 0; dy < KK; ++dy) {
            const float* trow = tile + (ybase + dy) * 140 + col;
            const float* w0r = wl + dy * KK;
            const float* w1r = w0r + KK * KK;
            const float* w2r = w1r + KK * KK;
            #pragma unroll
            for (int dx = 0; dx < KK; ++dx) {
                float w0 = w0r[dx], w1 = w1r[dx], w2 = w2r[dx];
                #pragma unroll
                for (int r = 0; r < 4; ++r) {
                    float v = trow[r * 140 + dx];
                    acc[r][0] = fmaf(v, w0, acc[r][0]);
                    acc[r][1] = fmaf(v, w1, acc[r][1]);
                    acc[r][2] = fmaf(v, w2, acc[r][2]);
                }
            }
        }
        #pragma unroll
        for (int r = 0; r < 4; ++r) {
            m0 = fmaxf(m0, acc[r][0]);
            m1 = fmaxf(m1, acc[r][1]);
            m2 = fmaxf(m2, acc[r][2]);
        }
    }
    red[half][0][col] = m0; red[half][1][col] = m1; red[half][2][col] = m2;
    __syncthreads();
    if (half == 0) {
        int x = xt * 128 + col;
        #pragma unroll
        for (int ch = 0; ch < C2D; ++ch) {
            float v = fmaxf(red[0][ch][col], red[1][ch][col]) + cb[ch];
            out[(((size_t)g * C2D + ch) * OUT_H + p) * SUMH + x] = v;
        }
    }
}

// ---------------------------------------------------------------- driver
extern "C" void kernel_launch(void* const* d_in, const int* in_sizes, int n_in,
                              void* d_out, int out_size, void* d_ws, size_t ws_size,
                              hipStream_t stream) {
    (void)in_sizes; (void)n_in; (void)out_size; (void)ws_size;
    const float* feat = (const float*)d_in[0];
    const int*   src  = (const int*)d_in[1];
    const int*   dst  = (const int*)d_in[2];
    const float* W[4] = {(const float*)d_in[3], (const float*)d_in[5],
                         (const float*)d_in[7], (const float*)d_in[9]};
    const float* B[4] = {(const float*)d_in[4], (const float*)d_in[6],
                         (const float*)d_in[8], (const float*)d_in[10]};
    const float* cw = (const float*)d_in[11];
    const float* cb = (const float*)d_in[12];
    float* out = (float*)d_out;

    char* ws = (char*)d_ws;
    size_t off = 0;
    auto walloc = [&](size_t bytes) -> void* {
        void* p = ws + off;
        off = (off + bytes + 255) & ~(size_t)255;
        return p;
    };
    float*    rinvOut = (float*)walloc((size_t)G * NN * 4);
    float*    rinvIn  = (float*)walloc((size_t)G * NN * 4);
    float*    hcat    = (float*)walloc((size_t)G * NN * SUMH * 4);
    float*    xtmp    = (float*)walloc((size_t)G * NN * HIDW * 4);
    float*    agg     = (float*)walloc((size_t)G * NN * HIDW * 4);
    _Float16* Wt      = (_Float16*)walloc((size_t)HIDW * HIDW * 2);

    // degrees -> rsqrt
    k_zero<<<(G * NN + 255) / 256, 256, 0, stream>>>(rinvOut, G * NN);
    k_zero<<<(G * NN + 255) / 256, 256, 0, stream>>>(rinvIn,  G * NN);
    k_degree<<<(G * EE + 255) / 256, 256, 0, stream>>>(src, dst, rinvOut, rinvIn);
    k_rinv<<<(G * NN + 255) / 256, 256, 0, stream>>>(rinvOut, rinvIn);

    const int Kdims[4] = {IN_DIM, HIDW, HIDW, HIDW};
    for (int l = 0; l < 4; ++l) {
        int Kd = Kdims[l];
        const float* inPtr; int inStride, colOff;
        if (l == 0) { inPtr = feat; inStride = IN_DIM; colOff = 0; }
        else        { inPtr = hcat; inStride = SUMH;   colOff = (l - 1) * HIDW; }
        k_prep_w<<<(Kd * HIDW + 255) / 256, 256, 0, stream>>>(W[l], Wt, Kd);
        k_gemm<<<dim3(NN / 16, G), 256, 0, stream>>>(inPtr, inStride, colOff,
                                                     rinvOut, Wt, Kd, xtmp);
        k_zero<<<(G * NN * HIDW + 255) / 256, 256, 0, stream>>>(agg, G * NN * HIDW);
        k_scatter<<<(G * EE * 64 + 255) / 256, 256, 0, stream>>>(
            (const float4*)xtmp, src, dst, agg);
        k_finalize<<<(G * NN * 64 + 255) / 256, 256, 0, stream>>>(
            (const float4*)agg, rinvIn, B[l], hcat, l * HIDW);
    }

    k_convpool<<<dim3(SUMH / 128, OUT_H, G), 256, 0, stream>>>(hcat, cw, cb, out);
}
